// HBMA_CUDA_Fused_83107617178061
// MI455X (gfx1250) — compile-verified, matrix operand order checked by reading, not run on hardware
//
#include <hip/hip_runtime.h>

typedef __attribute__((ext_vector_type(16))) _Float16 v16h;
typedef __attribute__((ext_vector_type(2)))  _Float16 h2;
typedef __attribute__((ext_vector_type(8)))  float    v8f;

#define N_    8
#define C_    3
#define H_    720
#define W_    1280
#define BLK   8
#define SRCH  4
#define BH    (H_/BLK)           // 90
#define BW    (W_/BLK)           // 160
#define TBY   2                  // block-rows per tile
#define TBX   8                  // block-cols per tile
#define TILE_H (TBY*BLK)         // 16
#define TILE_W (TBX*BLK)         // 64
#define REF_H  (TILE_H + 2*SRCH) // 24
#define REF_W  (TILE_W + 2*SRCH) // 72
#define REF_P  73                // f32 ref LDS pitch
#define REF_PH 74                // f16 ref LDS pitch (even => aligned pair loads)
#define PLANE  (C_*REF_H*REF_PH) // 5328 halves per f16 ref plane
#define TGT_PH 66                // f16 target LDS pitch (even)
#define NOFF   81                // (2*SRCH+1)^2
#define MV_ELEMS (N_*2*BH*BW)    // 230400

__global__ __launch_bounds__(256) void hbma_mv_zero(float* __restrict__ out) {
    int i = blockIdx.x * 256 + threadIdx.x;
    if (i < MV_ELEMS) out[i] = 0.0f;
}

__global__ __launch_bounds__(256) void hbma_sad_wmma(const float* __restrict__ ref,
                                                     const float* __restrict__ tgt,
                                                     float* __restrict__ out) {
    __shared__ float    sRef[C_ * REF_H * REF_P];    // 21,024 B f32: exact gather source
    __shared__ _Float16 sRefH[2 * PLANE];            // 21,312 B f16: plane0 = x, plane1 = x+1
    __shared__ _Float16 sTgtH[C_ * TILE_H * TGT_PH]; //  6,336 B f16: SAD operand
    __shared__ int      sBest[TBY * TBX];

    const int tid = threadIdx.x;
    const int n   = blockIdx.z;
    const int ty0 = blockIdx.y * TILE_H;
    const int tx0 = blockIdx.x * TILE_W;

    // ---- Stage reference tile (zero-padded +/-4 halo): f32 + dual-plane f16 ----
    for (int e = tid; e < C_ * REF_H * REF_W; e += 256) {
        int c  = e / (REF_H * REF_W);
        int r  = e % (REF_H * REF_W);
        int ry = r / REF_W, rx = r % REF_W;
        int gy = ty0 + ry - SRCH, gx = tx0 + rx - SRCH;
        float v = 0.0f;
        if (gy >= 0 && gy < H_ && gx >= 0 && gx < W_)
            v = ref[((size_t)(n * C_ + c) * H_ + gy) * W_ + gx];
        sRef[(c * REF_H + ry) * REF_P + rx] = v;
        int hb = (c * REF_H + ry) * REF_PH + rx;
        _Float16 hv = (_Float16)v;
        sRefH[hb] = hv;                              // plane0: logical x
        if (rx >= 1) sRefH[PLANE + hb - 1] = hv;     // plane1[i] = logical x = i+1
    }
    // ---- Stage target tile into LDS as f16 ----
    for (int e = tid; e < C_ * TILE_H * TILE_W; e += 256) {
        int c  = e / (TILE_H * TILE_W);
        int r  = e % (TILE_H * TILE_W);
        int ly = r / TILE_W, lx = r % TILE_W;
        float v = tgt[((size_t)(n * C_ + c) * H_ + (ty0 + ly)) * W_ + (tx0 + lx)];
        sTgtH[(c * TILE_H + ly) * TGT_PH + lx] = (_Float16)v;
    }
    __syncthreads();

    const int wave  = tid >> 5;
    const int lane  = tid & 31;
    const int mrow  = lane & 15;              // A-matrix row this lane feeds
    const int khalf = (lane & 16) ? 8 : 0;    // K-subset select per A layout

    v16h bones;
    #pragma unroll
    for (int j = 0; j < 16; ++j) bones[j] = (_Float16)1.0f;

    // Per-round ref-plane offsets (lane-dependent only; block/chunk invariant).
    // Odd dx reads the shifted plane at x-1 so every f16 pair load is 4B-aligned.
    int roff[6];
    #pragma unroll
    for (int round = 0; round < 6; ++round) {
        int oidx = round * 16 + mrow;          // candidate index (dy-outer, dx-inner)
        int o    = (oidx <= 80) ? oidx : 0;    // garbage rows clamped, excluded later
        int dy   = o / 9 - SRCH;
        int dx   = o % 9 - SRCH;
        roff[round] = dy * REF_PH + ((dx & 1) ? (PLANE + dx - 1) : dx);
    }

    // Each of the 8 waves owns 2 blocks of the 2x8 block tile.
    #pragma unroll
    for (int bi = 0; bi < 2; ++bi) {
        const int b  = wave * 2 + bi;
        const int br = b >> 3, bc = b & 7;
        const int by = br * BLK, bx = bc * BLK;

        v8f acc[6];
        #pragma unroll
        for (int r = 0; r < 6; ++r) { v8f z = {}; acc[r] = z; }

        for (int chunk = 0; chunk < 6; ++chunk) {
            // This lane's 16 K-slots: pixel p = chunk*32 + khalf + (j<8? j : j+8).
            // Pairs (2jp, 2jp+1) are horizontally adjacent (px even) => aligned h2 loads.
            h2  tp[8];
            int pbase[8];   // f16 ref plane0 index of even element of each pair
            #pragma unroll
            for (int jp = 0; jp < 8; ++jp) {
                int j   = jp * 2;
                int p   = chunk * 32 + khalf + (j < 8 ? j : j + 8);
                int c   = p >> 6;
                int rem = p & 63;
                int py  = rem >> 3, px = rem & 7;
                tp[jp]    = *(const h2*)&sTgtH[(c * TILE_H + by + py) * TGT_PH + bx + px];
                pbase[jp] = (c * REF_H + by + py + SRCH) * REF_PH + bx + px + SRCH;
            }
            #pragma unroll
            for (int round = 0; round < 6; ++round) {
                const int ro = roff[round];
                v16h a;
                #pragma unroll
                for (int jp = 0; jp < 8; ++jp) {
                    h2 rp = *(const h2*)&sRefH[pbase[jp] + ro];  // aligned ds_load_b32
                    h2 d  = rp - tp[jp];                         // v_pk_add_f16 (neg)
                    h2 ad = __builtin_elementwise_abs(d);        // v_and_b32 0x7fff7fff
                    a[jp * 2]     = ad[0];
                    a[jp * 2 + 1] = ad[1];
                }
                // D[M][*] += rowsum(|diff|) via ones-B  (f32 accumulate)
                acc[round] = __builtin_amdgcn_wmma_f32_16x16x32_f16(
                    false, a, false, bones, (short)0, acc[round], false, false);
            }
        }

        // Per-lane argmin over its D rows: M = r + 8*(lane>=16); ascending idx => first-win ties.
        float bestS = 3.0e38f;
        int   bestI = 1 << 20;
        const int mAdd = (lane & 16) ? 8 : 0;
        #pragma unroll
        for (int round = 0; round < 6; ++round) {
            #pragma unroll
            for (int r = 0; r < 8; ++r) {
                int idx = round * 16 + r + mAdd;
                float s = acc[round][r];
                if (idx < NOFF && s < bestS) { bestS = s; bestI = idx; }
            }
        }
        // Cross-lane argmin (wave32), tie-break on smallest candidate index.
        #pragma unroll
        for (int m = 16; m >= 1; m >>= 1) {
            float oS = __shfl_xor(bestS, m, 32);
            int   oI = __shfl_xor(bestI, m, 32);
            if (oS < bestS || (oS == bestS && oI < bestI)) { bestS = oS; bestI = oI; }
        }
        if (lane == 0) sBest[b] = bestI;
    }
    __syncthreads();

    // ---- Predicted frame: gather best-offset reference block from the f32 LDS halo tile ----
    float* pred = out + MV_ELEMS;
    for (int q = tid; q < C_ * TILE_H * TILE_W; q += 256) {
        int c   = q / (TILE_H * TILE_W);
        int r   = q % (TILE_H * TILE_W);
        int ly  = r / TILE_W, lx = r % TILE_W;
        int blk = (ly >> 3) * TBX + (lx >> 3);
        int idx = sBest[blk];
        int dy  = idx / 9 - SRCH;
        int dx  = idx % 9 - SRCH;
        float v = sRef[(c * REF_H + ly + SRCH + dy) * REF_P + lx + SRCH + dx];
        pred[((size_t)(n * C_ + c) * H_ + (ty0 + ly)) * W_ + (tx0 + lx)] = v;
    }
}

extern "C" void kernel_launch(void* const* d_in, const int* in_sizes, int n_in,
                              void* d_out, int out_size, void* d_ws, size_t ws_size,
                              hipStream_t stream) {
    const float* ref = (const float*)d_in[0];
    const float* tgt = (const float*)d_in[1];
    float* out = (float*)d_out;

    hbma_mv_zero<<<(MV_ELEMS + 255) / 256, 256, 0, stream>>>(out);

    dim3 grid(BW / TBX, BH / TBY, N_);   // 20 x 45 x 8 workgroups, 8 waves each
    hbma_sad_wmma<<<grid, 256, 0, stream>>>(ref, tgt, out);
}
